// MultiAttention_78417512890911
// MI455X (gfx1250) — matrix operand order repaired and checked by reading, not compiled, not run on hardware
//
#include <hip/hip_runtime.h>
#include <hip/hip_fp16.h>

typedef __attribute__((ext_vector_type(16))) _Float16 v16h;
typedef __attribute__((ext_vector_type(8)))  _Float16 v8h;
typedef __attribute__((ext_vector_type(8)))  float    v8f;
typedef __attribute__((ext_vector_type(4)))  unsigned int v4u;
typedef __attribute__((ext_vector_type(4)))  int v4i;
typedef __attribute__((ext_vector_type(8)))  int v8i;

#define NHEADS 2
#define DIN    100
#define KDIM   50
#define NB     8
#define NT     2048
#define NROWS  (NB*NT)      // 16384
#define KP     64           // padded per-head dim
#define CP     128          // padded model dim (2*KP)
#define WCOLS  112          // padded output cols (7*16)
#define KLDS_STRIDE 72      // halfs per K row in LDS (64 + 4-dword TDM pad)
#define KLDS_TILE   (32 * KLDS_STRIDE)

// ---------------- prep kernels ----------------

__global__ void prep_xh(const float* __restrict__ x, _Float16* __restrict__ xh) {
  int i = blockIdx.x * blockDim.x + threadIdx.x;
  if (i >= NROWS * CP) return;
  int r = i >> 7, c = i & 127;
  float v = (c < DIN) ? x[(size_t)r * DIN + c] : 0.0f;
  xh[i] = (_Float16)v;
}

__global__ void prep_w(const float* __restrict__ Wq, const float* __restrict__ Wk,
                       const float* __restrict__ Wv, const float* __restrict__ Wo,
                       _Float16* __restrict__ WqT, _Float16* __restrict__ WkT,
                       _Float16* __restrict__ WvT, _Float16* __restrict__ WoT) {
  int i = blockIdx.x * blockDim.x + threadIdx.x;
  if (i >= WCOLS * CP) return;
  int c = i >> 7, kk = i & 127;
  float q = 0.f, k2 = 0.f, v = 0.f, o = 0.f;
  if (c < DIN && kk < DIN) {
    q  = Wq[(size_t)kk * DIN + c] * 0.1f;   // fold both 1/SCALE factors into Wq
    k2 = Wk[(size_t)kk * DIN + c];
    v  = Wv[(size_t)kk * DIN + c];
  }
  int h = kk >> 6, w = kk & 63;
  if (c < DIN && w < KDIM) o = Wo[(size_t)(h * KDIM + w) * DIN + c];
  WqT[i] = (_Float16)q; WkT[i] = (_Float16)k2; WvT[i] = (_Float16)v; WoT[i] = (_Float16)o;
}

// ---------------- TDM helper: async 32x64-f16 tile -> LDS (padded rows) -------
__device__ __forceinline__ void tdm_load_k32x64(unsigned int lds_off,
                                                const _Float16* gptr) {
  unsigned long long ga = (unsigned long long)(size_t)gptr;
  v4u g0;
  g0[0] = 1u;                                         // count=1, user mode, no gather
  g0[1] = lds_off;                                    // lds_addr (bytes)
  g0[2] = (unsigned int)(ga & 0xffffffffu);           // global_addr[31:0]
  g0[3] = (unsigned int)((ga >> 32) & 0x01ffffffu)    // global_addr[56:32]
          | (2u << 30);                               // type = 2 ("image")
  v8i g1;
  g1[0] = (1 << 16)        // data_size = 1 -> 2 bytes
        | (1 << 20)        // pad_enable
        | (4 << 22)        // pad_interval: 32 DWORDs (one 128B row)
        | (3 << 25);       // pad_amount: 4 DWORDs -> 144B row stride in LDS
  g1[1] = (64 << 16);      // tensor_dim0 = 64 (low 16 bits at [63:48])
  g1[2] = (32 << 16);      // tensor_dim0 hi = 0 | tensor_dim1 = 32
  g1[3] = (64 << 16);      // tensor_dim1 hi = 0 | tile_dim0 = 64
  g1[4] = 32;              // tile_dim1 = 32 | tile_dim2 = 0
  g1[5] = 64;              // tensor_dim0_stride = 64 (data_size units)
  g1[6] = 0;
  g1[7] = 0;
  v4i z4 = {0, 0, 0, 0};
#if defined(__clang_major__) && (__clang_major__ >= 23)
  v8i z8 = {0, 0, 0, 0, 0, 0, 0, 0};
  __builtin_amdgcn_tensor_load_to_lds(g0, g1, z4, z4, z8, 0);
#else
  __builtin_amdgcn_tensor_load_to_lds(g0, g1, z4, z4, 0);
#endif
}

// ---------------- projection GEMM: Q/K/V = xh @ W ----------------
__global__ __launch_bounds__(32)
void proj_kernel(const _Float16* __restrict__ xh,
                 const _Float16* __restrict__ WqT, const _Float16* __restrict__ WkT,
                 const _Float16* __restrict__ WvT,
                 _Float16* __restrict__ Qh, _Float16* __restrict__ Kh,
                 _Float16* __restrict__ VhT) {
  const int lane = threadIdx.x & 31;
  const int lm = lane & 15, hl = lane >> 4;
  const int tbase = blockIdx.x * 16;
  const int ctile = blockIdx.y;
  const int z = blockIdx.z;
  const _Float16* WT = (z == 0) ? WqT : (z == 1) ? WkT : WvT;

  const _Float16* arow = xh + (size_t)(tbase + lm) * CP;
  const _Float16* brow = WT + (size_t)(ctile * 16 + lm) * CP;

  v8f acc = {};
#pragma unroll
  for (int s = 0; s < 4; ++s) {
    v8h a0 = *(const v8h*)(arow + 32 * s + 16 * hl);
    v8h a1 = *(const v8h*)(arow + 32 * s + 16 * hl + 8);
    v16h a;
#pragma unroll
    for (int e = 0; e < 8; ++e) { a[e] = a0[e]; a[8 + e] = a1[e]; }
    v16h b = *(const v16h*)(brow + 32 * s + 16 * hl);
    acc = __builtin_amdgcn_wmma_f32_16x16x32_f16(false, a, false, b, (short)0, acc,
                                                 false, false);
  }
  int c = ctile * 16 + lm;
  if (c < DIN) {
    int h = c / KDIM, cc = c % KDIM;
#pragma unroll
    for (int j = 0; j < 8; ++j) {
      int row = tbase + j + 8 * hl;
      int bi = row >> 11, tt = row & 2047;
      int bh = bi * 2 + h;
      _Float16 val = (_Float16)acc[j];
      if (z == 0)      Qh[((size_t)bh * NT + tt) * KP + cc] = val;
      else if (z == 1) Kh[((size_t)bh * NT + tt) * KP + cc] = val;
      else             VhT[((size_t)bh * KP + cc) * NT + tt] = val;  // V transposed
    }
  }
}

// ---------------- flash attention: one wave per (bh, 32-query tile) ----------
__global__ __launch_bounds__(32)
void attn_kernel(const _Float16* __restrict__ Qh, const _Float16* __restrict__ Kh,
                 const _Float16* __restrict__ VhT, const float* __restrict__ mask,
                 _Float16* __restrict__ ctx) {
  __shared__ __align__(16) _Float16 kbuf[2 * KLDS_TILE];  // TDM double buffer
  __shared__ __align__(16) _Float16 pbuf[32 * 40];        // 32x32 P tile, padded
  const int lane = threadIdx.x & 31;
  const int lm = lane & 15, hl = lane >> 4;
  const int bh = blockIdx.x;            // 0..15
  const int qbase = blockIdx.y * 32;    // 32 queries per wave
  const int bb = bh >> 1, hh = bh & 1;
  const float* mrow = mask + (size_t)(bh & 7) * NT;  // concat([mask,mask]) remap

  const _Float16* Kb = Kh + (size_t)bh * NT * KP;
  const _Float16* Vb = VhT + (size_t)bh * KP * NT;
  unsigned int koff[2];
  koff[0] = (unsigned int)(size_t)(&kbuf[0]);
  koff[1] = (unsigned int)(size_t)(&kbuf[KLDS_TILE]);

  // Q A-fragments for both 16-query subtiles, kept in registers throughout
  v16h aq[2][2];
#pragma unroll
  for (int qi = 0; qi < 2; ++qi) {
    const _Float16* qrow = Qh + ((size_t)bh * NT + qbase + qi * 16 + lm) * KP;
#pragma unroll
    for (int s = 0; s < 2; ++s) {
      v8h a0 = *(const v8h*)(qrow + 32 * s + 16 * hl);
      v8h a1 = *(const v8h*)(qrow + 32 * s + 16 * hl + 8);
#pragma unroll
      for (int e = 0; e < 8; ++e) { aq[qi][s][e] = a0[e]; aq[qi][s][8 + e] = a1[e]; }
    }
  }

  v8f O[2][4] = {};
  float mrun[2][8], lrun[2][8];
#pragma unroll
  for (int qi = 0; qi < 2; ++qi)
#pragma unroll
    for (int j = 0; j < 8; ++j) { mrun[qi][j] = -__builtin_inff(); lrun[qi][j] = 0.0f; }

  // prime the TDM pipeline with key tile 0
  tdm_load_k32x64(koff[0], Kb);

  for (int kb = 0; kb < NT; kb += 32) {
    const int cur = (kb >> 5) & 1;
    if (kb + 32 < NT) {
      tdm_load_k32x64(koff[cur ^ 1], Kb + (size_t)(kb + 32) * KP);
      __builtin_amdgcn_s_wait_tensorcnt(1);   // current tile complete, next in flight
    } else {
      __builtin_amdgcn_s_wait_tensorcnt(0);
    }
    asm volatile("" ::: "memory");            // keep LDS reads below the wait
    const _Float16* kt = &kbuf[cur * KLDS_TILE];

    // prefetch next V tile (rows striped across lanes; one cacheline each)
    if (kb + 32 < NT) {
      __builtin_prefetch(Vb + (size_t)(lane * 2) * NT + kb + 32, 0, 3);
      __builtin_prefetch(Vb + (size_t)(lane * 2 + 1) * NT + kb + 32, 0, 3);
    }

    // S = Q @ K^T : 32 queries x 32 keys; B fragments reused by both query tiles
    v8f S[2][2] = {};
#pragma unroll
    for (int g = 0; g < 2; ++g) {
#pragma unroll
      for (int s = 0; s < 2; ++s) {
        v16h bf = *(const v16h*)(kt + (g * 16 + lm) * KLDS_STRIDE + 32 * s + 16 * hl);
        S[0][g] = __builtin_amdgcn_wmma_f32_16x16x32_f16(false, aq[0][s], false, bf,
                                                         (short)0, S[0][g], false, false);
        S[1][g] = __builtin_amdgcn_wmma_f32_16x16x32_f16(false, aq[1][s], false, bf,
                                                         (short)0, S[1][g], false, false);
      }
    }
    float madd0 = mrow[kb + lm];
    float madd1 = mrow[kb + 16 + lm];

    __syncthreads();
#pragma unroll
    for (int qi = 0; qi < 2; ++qi) {
      float p0[8], p1[8];
#pragma unroll
      for (int j = 0; j < 8; ++j) { p0[j] = S[qi][0][j] + madd0; p1[j] = S[qi][1][j] + madd1; }
#pragma unroll
      for (int j = 0; j < 8; ++j) {
        float tm = fmaxf(p0[j], p1[j]);
        tm = fmaxf(tm, __shfl_xor(tm, 1));
        tm = fmaxf(tm, __shfl_xor(tm, 2));
        tm = fmaxf(tm, __shfl_xor(tm, 4));
        tm = fmaxf(tm, __shfl_xor(tm, 8));
        float mn = fmaxf(mrun[qi][j], tm);
        float sc = __expf(mrun[qi][j] - mn);
        p0[j] = __expf(p0[j] - mn);
        p1[j] = __expf(p1[j] - mn);
        float rs = p0[j] + p1[j];
        rs += __shfl_xor(rs, 1);
        rs += __shfl_xor(rs, 2);
        rs += __shfl_xor(rs, 4);
        rs += __shfl_xor(rs, 8);
        lrun[qi][j] = lrun[qi][j] * sc + rs;
        mrun[qi][j] = mn;
#pragma unroll
        for (int nt = 0; nt < 4; ++nt) O[qi][nt][j] *= sc;
      }
#pragma unroll
      for (int j = 0; j < 8; ++j) {
        pbuf[(qi * 16 + j + 8 * hl) * 40 + lm]      = (_Float16)p0[j];
        pbuf[(qi * 16 + j + 8 * hl) * 40 + 16 + lm] = (_Float16)p1[j];
      }
    }
    __syncthreads();
    v16h ap[2];
#pragma unroll
    for (int qi = 0; qi < 2; ++qi) {
      v8h x0 = *(const v8h*)(&pbuf[(qi * 16 + lm) * 40 + 16 * hl]);
      v8h x1 = *(const v8h*)(&pbuf[(qi * 16 + lm) * 40 + 16 * hl + 8]);
#pragma unroll
      for (int e = 0; e < 8; ++e) { ap[qi][e] = x0[e]; ap[qi][8 + e] = x1[e]; }
    }

    // O += P @ V ; each V B-fragment serves both query tiles
#pragma unroll
    for (int nt = 0; nt < 4; ++nt) {
      v16h bv = *(const v16h*)(Vb + (size_t)(nt * 16 + lm) * NT + kb + 16 * hl);
      O[0][nt] = __builtin_amdgcn_wmma_f32_16x16x32_f16(false, ap[0], false, bv,
                                                        (short)0, O[0][nt], false, false);
      O[1][nt] = __builtin_amdgcn_wmma_f32_16x16x32_f16(false, ap[1], false, bv,
                                                        (short)0, O[1][nt], false, false);
    }
  }

  // epilogue: normalize and store ctx (padded cols are 0 since V pads are 0)
#pragma unroll
  for (int qi = 0; qi < 2; ++qi) {
#pragma unroll
    for (int j = 0; j < 8; ++j) {
      float inv = 1.0f / lrun[qi][j];
      int row = qbase + qi * 16 + j + 8 * hl;
      size_t base = ((size_t)(bb * NT + row)) * CP + hh * KP;
#pragma unroll
      for (int nt = 0; nt < 4; ++nt)
        ctx[base + nt * 16 + lm] = (_Float16)(O[qi][nt][j] * inv);
    }
  }
}

// ---------------- output GEMM: out = x + ctx @ Wo + bo ----------------
__global__ __launch_bounds__(32)
void out_kernel(const _Float16* __restrict__ ctx, const _Float16* __restrict__ WoT,
                const float* __restrict__ x, const float* __restrict__ bo,
                float* __restrict__ out) {
  const int lane = threadIdx.x & 31;
  const int lm = lane & 15, hl = lane >> 4;
  const int tbase = blockIdx.x * 16;
  const int ctile = blockIdx.y;
  const _Float16* arow = ctx + (size_t)(tbase + lm) * CP;
  const _Float16* brow = WoT + (size_t)(ctile * 16 + lm) * CP;

  v8f acc = {};
#pragma unroll
  for (int s = 0; s < 4; ++s) {
    v8h a0 = *(const v8h*)(arow + 32 * s + 16 * hl);
    v8h a1 = *(const v8h*)(arow + 32 * s + 16 * hl + 8);
    v16h a;
#pragma unroll
    for (int e = 0; e < 8; ++e) { a[e] = a0[e]; a[8 + e] = a1[e]; }
    v16h b = *(const v16h*)(brow + 32 * s + 16 * hl);
    acc = __builtin_amdgcn_wmma_f32_16x16x32_f16(false, a, false, b, (short)0, acc,
                                                 false, false);
  }
  int c = ctile * 16 + lm;
  if (c < DIN) {
    float bias = bo[c];
#pragma unroll
    for (int j = 0; j < 8; ++j) {
      int row = tbase + j + 8 * hl;
      out[(size_t)row * DIN + c] = x[(size_t)row * DIN + c] + acc[j] + bias;
    }
  }
}

// ---------------- launch ----------------

extern "C" void kernel_launch(void* const* d_in, const int* in_sizes, int n_in,
                              void* d_out, int out_size, void* d_ws, size_t ws_size,
                              hipStream_t stream) {
  const float* x    = (const float*)d_in[0];
  const float* mask = (const float*)d_in[1];
  const float* Wq   = (const float*)d_in[2];
  const float* Wk   = (const float*)d_in[3];
  const float* Wv   = (const float*)d_in[4];
  const float* Wo   = (const float*)d_in[5];
  const float* bo   = (const float*)d_in[6];
  float* out = (float*)d_out;

  char* ws = (char*)d_ws;
  const size_t SZ_XH  = (size_t)NROWS * CP * 2;        // 4 MB
  const size_t SZ_QKV = (size_t)16 * NT * KP * 2;      // 4 MB each
  const size_t SZ_CTX = (size_t)NROWS * CP * 2;        // 4 MB
  const size_t SZ_W   = (size_t)WCOLS * CP * 2;        // 28 KB each
  _Float16* xh  = (_Float16*)(ws);
  _Float16* Qh  = (_Float16*)(ws + SZ_XH);
  _Float16* Kh  = (_Float16*)(ws + SZ_XH + SZ_QKV);
  _Float16* VhT = (_Float16*)(ws + SZ_XH + 2 * SZ_QKV);
  _Float16* ctx = (_Float16*)(ws + SZ_XH + 3 * SZ_QKV);
  _Float16* WqT = (_Float16*)(ws + SZ_XH + 3 * SZ_QKV + SZ_CTX);
  _Float16* WkT = (_Float16*)(ws + SZ_XH + 3 * SZ_QKV + SZ_CTX + SZ_W);
  _Float16* WvT = (_Float16*)(ws + SZ_XH + 3 * SZ_QKV + SZ_CTX + 2 * SZ_W);
  _Float16* WoT = (_Float16*)(ws + SZ_XH + 3 * SZ_QKV + SZ_CTX + 3 * SZ_W);

  // zero Q/K/V so padded dims (50..63) contribute nothing
  hipMemsetAsync(Qh, 0, SZ_QKV, stream);
  hipMemsetAsync(Kh, 0, SZ_QKV, stream);
  hipMemsetAsync(VhT, 0, SZ_QKV, stream);

  prep_xh<<<(NROWS * CP + 255) / 256, 256, 0, stream>>>(x, xh);
  prep_w<<<(WCOLS * CP + 255) / 256, 256, 0, stream>>>(Wq, Wk, Wv, Wo, WqT, WkT, WvT, WoT);

  proj_kernel<<<dim3(NROWS / 16, WCOLS / 16, 3), 32, 0, stream>>>(
      xh, WqT, WkT, WvT, Qh, Kh, VhT);

  attn_kernel<<<dim3(16, NT / 32), 32, 0, stream>>>(Qh, Kh, VhT, mask, ctx);

  out_kernel<<<dim3(NROWS / 16, WCOLS / 16), 32, 0, stream>>>(ctx, WoT, x, bo, out);
}
